// RNN_28518582845480
// MI455X (gfx1250) — matrix operand order repaired and checked by reading, not compiled, not run on hardware
//
#include <hip/hip_runtime.h>
#include <hip/hip_bf16.h>

// ---------------------------------------------------------------------------
// 2-layer LSTM + vocab head, T=20 steps.
// - bf16 WMMA (v_wmma_f32_16x16x32_bf16), fp32 accumulation, fp32 cell state.
// - Weights converted once per launch to padded bf16 (L2-resident, ~58MB).
// - Async global->LDS (ASYNCcnt) double-buffered staging, b128 per lane.
// - Deterministic split-K (partials + fused reduction in the LSTM pointwise
//   kernel) to raise workgroup count on the dominant 128x2048x10048 GEMM.
// ---------------------------------------------------------------------------

typedef __attribute__((ext_vector_type(16))) __bf16 v16bf;
typedef __attribute__((ext_vector_type(8)))  __bf16 v8bf;
typedef __attribute__((ext_vector_type(8)))  float  v8f;

#define LSTM_B 128
#define LSTM_D 512
#define LSTM_V 10000
#define LSTM_T 20
#define LSTM_G 2048          // 4*D gate width
#define VPAD   10048         // V padded to multiple of 64 (157*64)

#define BM 64
#define BN 64
#define BK 64                // two 16x16x32 K-slices per stage
#define LDSS 72              // LDS row stride in bf16 (144B, 16B-aligned frags)

#define USE_ASYNC_LDS 1      // gfx1250 async global->LDS path

__device__ __forceinline__ unsigned short f2bf(float f) {
  unsigned int u = __float_as_uint(f);
  u += 0x7FFFu + ((u >> 16) & 1u);           // round-to-nearest-even
  return (unsigned short)(u >> 16);
}

__device__ __forceinline__ float sigmoidf_(float x) {
  return 1.0f / (1.0f + __expf(-x));
}

// ---- gfx1250 async global->LDS helpers ------------------------------------
__device__ __forceinline__ void async_b128(unsigned int ldsOff, const void* g) {
#if USE_ASYNC_LDS
  asm volatile("global_load_async_to_lds_b128 %0, %1, off"
               :: "v"(ldsOff), "v"((unsigned long long)(size_t)g)
               : "memory");
#endif
}
__device__ __forceinline__ void wait_async_le8() {
#if USE_ASYNC_LDS
  asm volatile("s_wait_asynccnt 0x8" ::: "memory");
#endif
}
__device__ __forceinline__ void wait_async_0() {
#if USE_ASYNC_LDS
  asm volatile("s_wait_asynccnt 0x0" ::: "memory");
#endif
}

// Issue one 64x64 bf16 tile of X and of W into LDS buffers (8 b128 per thread).
__device__ __forceinline__ void stage_tiles(const unsigned short* __restrict__ X,
                                            const unsigned short* __restrict__ W,
                                            int ldX, int ldW, int mTile, int nTile,
                                            int kb, __bf16* xs, __bf16* ws, int tid) {
#pragma unroll
  for (int i = 0; i < 4; ++i) {
    int idx = tid + i * 128;                 // 0..511
    int row = idx >> 3;                      // 64 rows, 8 b128 per row
    int c   = (idx & 7) * 8;                 // bf16 column
    const void* gx = X + (size_t)(mTile + row) * ldX + kb + c;
    const void* gw = W + (size_t)(nTile + row) * ldW + kb + c;
#if USE_ASYNC_LDS
    async_b128((unsigned int)(size_t)(xs + row * LDSS + c), gx);
    async_b128((unsigned int)(size_t)(ws + row * LDSS + c), gw);
#else
    *reinterpret_cast<v8bf*>(xs + row * LDSS + c) =
        *reinterpret_cast<const v8bf*>(gx);
    *reinterpret_cast<v8bf*>(ws + row * LDSS + c) =
        *reinterpret_cast<const v8bf*>(gw);
#endif
  }
}

// A fragment 16x32: lanes 0-15 rows M=0..15 K {0..7,16..23}; lanes 16-31 same
// rows K {8..15,24..31}. Two 16B LDS loads.
__device__ __forceinline__ v16bf load_frag_a(const __bf16* s, int mBase, int kBase, int lane) {
  int m  = mBase + (lane & 15);
  int kh = (lane >> 4) << 3;
  const __bf16* p = s + m * LDSS + kBase + kh;
  union { v16bf v; v8bf h[2]; } u;
  u.h[0] = *reinterpret_cast<const v8bf*>(p);
  u.h[1] = *reinterpret_cast<const v8bf*>(p + 16);
  return u.v;
}

// B fragment 32x16: lane holds N = lane&15, K {0..7,16..23} (lanes 0-15) / {8..15,24..31} (lanes 16-31).
__device__ __forceinline__ v16bf load_frag_b(const __bf16* s, int nBase, int kBase, int lane) {
  int n  = nBase + (lane & 15);
  int ko = (lane >> 4) << 3;
  const __bf16* p = s + n * LDSS + kBase + ko;
  union { v16bf v; v8bf h[2]; } u;
  u.h[0] = *reinterpret_cast<const v8bf*>(p);
  u.h[1] = *reinterpret_cast<const v8bf*>(p + 16);
  return u.v;
}

// Y = X * W^T over a K-chunk selected by blockIdx.z (split-K).
//   z <  nChunks0 : operand set 0, K-slices [z*chunk0, z*chunk0+cnt) of slices0
//   z >= nChunks0 : operand set 1, all slices1 K-slices
// parts != nullptr: write fp32 partial slab z (M x 2048, no bias/guard).
// parts == nullptr: final mode: +bias, fp32 store (cols<Nreal) + bf16 store.
// K-slice unit = 64 bf16. All lds/pads multiples of 64; M,Npad multiples of 64.
__global__ __launch_bounds__(128)
void gemm_wmma_bf16(const unsigned short* __restrict__ X0,
                    const unsigned short* __restrict__ W0,
                    int ld0, int slices0, int chunk0, int nChunks0,
                    const unsigned short* __restrict__ X1,
                    const unsigned short* __restrict__ W1,
                    int ld1, int slices1,
                    float* __restrict__ parts,
                    const float* __restrict__ bias,
                    float* __restrict__ Yf, int ldYf, int Nreal,
                    unsigned short* __restrict__ Ybf, int ldYbf) {
  __shared__ __bf16 Xs[2][BM * LDSS];
  __shared__ __bf16 Ws[2][BN * LDSS];

  const int tid   = threadIdx.x;
  const int lane  = tid & 31;
  const int wave  = tid >> 5;                // 0..3
  const int wm    = (wave >> 1) * 32;
  const int wn    = (wave & 1) * 32;
  const int mTile = blockIdx.y * BM;
  const int nTile = blockIdx.x * BN;
  const int z     = blockIdx.z;

  const unsigned short* X;
  const unsigned short* W;
  int ld, kStart, kCnt;
  if (z < nChunks0) {
    X = X0; W = W0; ld = ld0;
    kStart = z * chunk0;
    int rem = slices0 - kStart;
    kCnt = rem < chunk0 ? rem : chunk0;
  } else {
    X = X1; W = W1; ld = ld1;
    kStart = 0;
    kCnt = slices1;
  }

  v8f acc[2][2] = {};

  // Double-buffered pipeline: async DMA of stage k+1 overlaps WMMA on stage k.
  int cur = 0;
  stage_tiles(X, W, ld, ld, mTile, nTile, kStart * BK, Xs[0], Ws[0], tid);
  for (int it = 0; it < kCnt; ++it) {
    if (it + 1 < kCnt) {
      stage_tiles(X, W, ld, ld, mTile, nTile, (kStart + it + 1) * BK,
                  Xs[cur ^ 1], Ws[cur ^ 1], tid);
      wait_async_le8();                      // current stage's 8 asyncs done
    } else {
      wait_async_0();
    }
    __syncthreads();                         // all waves staged buffer `cur`
#if !USE_ASYNC_LDS
    (void)0;
#endif
    const __bf16* xs = Xs[cur];
    const __bf16* ws = Ws[cur];
#pragma unroll
    for (int s = 0; s < 2; ++s) {            // two 16x16x32 K-slices
      v16bf a0 = load_frag_a(xs, wm,      s * 32, lane);
      v16bf a1 = load_frag_a(xs, wm + 16, s * 32, lane);
      v16bf b0 = load_frag_b(ws, wn,      s * 32, lane);
      v16bf b1 = load_frag_b(ws, wn + 16, s * 32, lane);
      acc[0][0] = __builtin_amdgcn_wmma_f32_16x16x32_bf16(false, a0, false, b0, (short)0, acc[0][0], false, false);
      acc[0][1] = __builtin_amdgcn_wmma_f32_16x16x32_bf16(false, a0, false, b1, (short)0, acc[0][1], false, false);
      acc[1][0] = __builtin_amdgcn_wmma_f32_16x16x32_bf16(false, a1, false, b0, (short)0, acc[1][0], false, false);
      acc[1][1] = __builtin_amdgcn_wmma_f32_16x16x32_bf16(false, a1, false, b1, (short)0, acc[1][1], false, false);
    }
    __syncthreads();                         // buffer `cur` free for next DMA
    cur ^= 1;
  }

  // C/D layout: vgpr r, lane -> M = r + 8*(lane>>4), N = lane&15.
#pragma unroll
  for (int i = 0; i < 2; ++i) {
#pragma unroll
    for (int j = 0; j < 2; ++j) {
      int n  = nTile + wn + j * 16 + (lane & 15);
      int m0 = mTile + wm + i * 16 + ((lane >> 4) << 3);
      if (parts) {                           // split-K partial slab z (N==2048)
        float* dst = parts + ((size_t)z * LSTM_B + m0) * LSTM_G + n;
#pragma unroll
        for (int r = 0; r < 8; ++r) dst[(size_t)r * LSTM_G] = acc[i][j][r];
      } else if (n < Nreal) {                // final: bias + fp32 + bf16 stores
        float bv = bias ? bias[n] : 0.0f;
#pragma unroll
        for (int r = 0; r < 8; ++r) {
          float v = acc[i][j][r] + bv;
          Yf[(size_t)(m0 + r) * ldYf + n] = v;
          if (Ybf) Ybf[(size_t)(m0 + r) * ldYbf + n] = f2bf(v);
        }
      }
    }
  }
}

// Fused split-K reduction + biases + LSTM pointwise (torch gate order i,f,g,o).
__global__ void lstm_act(const float* __restrict__ parts, int nParts,
                         const float* __restrict__ b_ih,
                         const float* __restrict__ b_hh,
                         float* __restrict__ c,
                         unsigned short* __restrict__ hb) {
  int idx = blockIdx.x * blockDim.x + threadIdx.x;   // 0 .. B*D-1
  if (idx >= LSTM_B * LSTM_D) return;
  int b = idx >> 9;
  int d = idx & (LSTM_D - 1);
  float g0 = b_ih[d]               + b_hh[d];
  float g1 = b_ih[LSTM_D + d]      + b_hh[LSTM_D + d];
  float g2 = b_ih[2 * LSTM_D + d]  + b_hh[2 * LSTM_D + d];
  float g3 = b_ih[3 * LSTM_D + d]  + b_hh[3 * LSTM_D + d];
  for (int p = 0; p < nParts; ++p) {
    const float* base = parts + ((size_t)p * LSTM_B + b) * LSTM_G + d;
    g0 += base[0];
    g1 += base[LSTM_D];
    g2 += base[2 * LSTM_D];
    g3 += base[3 * LSTM_D];
  }
  float cn = sigmoidf_(g1) * c[idx] + sigmoidf_(g0) * tanhf(g2);
  float hn = sigmoidf_(g3) * tanhf(cn);
  c[idx]  = cn;
  hb[idx] = f2bf(hn);
}

// fp32 (rows x cols) -> zero-padded bf16 (rowsPad x colsPad)
__global__ void convert_pad_bf16(const float* __restrict__ src,
                                 unsigned short* __restrict__ dst,
                                 int rows, int cols, int rowsPad, int colsPad) {
  size_t total = (size_t)rowsPad * colsPad;
  for (size_t i = (size_t)blockIdx.x * blockDim.x + threadIdx.x; i < total;
       i += (size_t)gridDim.x * blockDim.x) {
    int r = (int)(i / colsPad);
    int cc = (int)(i % colsPad);
    float v = (r < rows && cc < cols) ? src[(size_t)r * cols + cc] : 0.0f;
    dst[i] = f2bf(v);
  }
}

__global__ void zero_u16(unsigned short* __restrict__ p, size_t n) {
  for (size_t i = (size_t)blockIdx.x * blockDim.x + threadIdx.x; i < n;
       i += (size_t)gridDim.x * blockDim.x)
    p[i] = 0;
}

__global__ void init_state(const float* __restrict__ inp,
                           float* __restrict__ c0, float* __restrict__ c1,
                           unsigned short* __restrict__ h0b,
                           unsigned short* __restrict__ h1b) {
  int i = blockIdx.x * blockDim.x + threadIdx.x;
  if (i < LSTM_B * LSTM_D) {
    c0[i] = 0.0f;
    c1[i] = 0.0f;
    h0b[i] = f2bf(inp[i]);   // H[0] = image features
    h1b[i] = 0;              // bf16 zero
  }
}

extern "C" void kernel_launch(void* const* d_in, const int* in_sizes, int n_in,
                              void* d_out, int out_size, void* d_ws, size_t ws_size,
                              hipStream_t stream) {
  (void)in_sizes; (void)n_in; (void)out_size; (void)ws_size;

  const float* input_ = (const float*)d_in[0];
  const float* W_ih0  = (const float*)d_in[1];
  const float* W_hh0  = (const float*)d_in[2];
  const float* b_ih0  = (const float*)d_in[3];
  const float* b_hh0  = (const float*)d_in[4];
  const float* W_ih1  = (const float*)d_in[5];
  const float* W_hh1  = (const float*)d_in[6];
  const float* b_ih1  = (const float*)d_in[7];
  const float* b_hh1  = (const float*)d_in[8];
  const float* W_enc  = (const float*)d_in[9];
  const float* b_enc  = (const float*)d_in[10];
  float* out = (float*)d_out;

  // ---- workspace carve-up (256B aligned), ~71 MB ----
  unsigned char* ws = (unsigned char*)d_ws;
  size_t off = 0;
  auto carve = [&](size_t bytes) -> void* {
    void* p = ws + off;
    off = (off + bytes + 255) & ~(size_t)255;
    return p;
  };
  unsigned short* Wih0b = (unsigned short*)carve((size_t)LSTM_G * VPAD * 2);
  unsigned short* Whh0b = (unsigned short*)carve((size_t)LSTM_G * LSTM_D * 2);
  unsigned short* Wih1b = (unsigned short*)carve((size_t)LSTM_G * LSTM_D * 2);
  unsigned short* Whh1b = (unsigned short*)carve((size_t)LSTM_G * LSTM_D * 2);
  unsigned short* Wencb = (unsigned short*)carve((size_t)VPAD * LSTM_D * 2);
  unsigned short* prevb = (unsigned short*)carve((size_t)LSTM_B * VPAD * 2);
  unsigned short* h0b   = (unsigned short*)carve((size_t)LSTM_B * LSTM_D * 2);
  unsigned short* h1b   = (unsigned short*)carve((size_t)LSTM_B * LSTM_D * 2);
  float* parts = (float*)carve((size_t)9 * LSTM_B * LSTM_G * 4);  // split-K slabs
  float* c0    = (float*)carve((size_t)LSTM_B * LSTM_D * 4);
  float* c1    = (float*)carve((size_t)LSTM_B * LSTM_D * 4);

  // ---- one-time (per launch): bf16 weight copies + state init ----
  convert_pad_bf16<<<2048, 256, 0, stream>>>(W_ih0, Wih0b, LSTM_G, LSTM_V, LSTM_G, VPAD);
  convert_pad_bf16<<<1024, 256, 0, stream>>>(W_hh0, Whh0b, LSTM_G, LSTM_D, LSTM_G, LSTM_D);
  convert_pad_bf16<<<1024, 256, 0, stream>>>(W_ih1, Wih1b, LSTM_G, LSTM_D, LSTM_G, LSTM_D);
  convert_pad_bf16<<<1024, 256, 0, stream>>>(W_hh1, Whh1b, LSTM_G, LSTM_D, LSTM_G, LSTM_D);
  convert_pad_bf16<<<2048, 256, 0, stream>>>(W_enc, Wencb, LSTM_V, LSTM_D, VPAD, LSTM_D);
  zero_u16<<<2048, 256, 0, stream>>>(prevb, (size_t)LSTM_B * VPAD);
  init_state<<<(LSTM_B * LSTM_D + 255) / 256, 256, 0, stream>>>(input_, c0, c1, h0b, h1b);

  const int VSLICES = VPAD / BK;             // 157 K-slices of the vocab GEMM
  const int DSLICES = LSTM_D / BK;           // 8
  const int CHUNK0  = 20;                    // ceil(157/8): 7x20 + 17
  const dim3 gGates0(LSTM_G / BN, LSTM_B / BM, 9);   // 8 vocab chunks + hidden
  const dim3 gGates1(LSTM_G / BN, LSTM_B / BM, 2);   // two D-GEMMs as partials
  const dim3 gEnc(VPAD / BN, LSTM_B / BM, 1);
  const int actBlocks = (LSTM_B * LSTM_D + 255) / 256;

  // ---- T sequential steps (stream order provides the dependency chain) ----
  for (int t = 0; t < LSTM_T; ++t) {
    // layer 0: prev_logits*W_ih0^T (split-K x8) + h0*W_hh0^T -> 9 partials
    gemm_wmma_bf16<<<gGates0, 128, 0, stream>>>(
        prevb, Wih0b, VPAD, VSLICES, CHUNK0, 8,
        h0b, Whh0b, LSTM_D, DSLICES,
        parts, nullptr, nullptr, 0, 0, nullptr, 0);
    lstm_act<<<actBlocks, 256, 0, stream>>>(parts, 9, b_ih0, b_hh0, c0, h0b);

    // layer 1: h0*W_ih1^T + h1*W_hh1^T -> 2 partials
    gemm_wmma_bf16<<<gGates1, 128, 0, stream>>>(
        h0b, Wih1b, LSTM_D, DSLICES, DSLICES, 1,
        h1b, Whh1b, LSTM_D, DSLICES,
        parts, nullptr, nullptr, 0, 0, nullptr, 0);
    lstm_act<<<actBlocks, 256, 0, stream>>>(parts, 2, b_ih1, b_hh1, c1, h1b);

    // logits: h1*W_enc^T + b_enc -> d_out[t] (fp32) and prevb (bf16 feedback)
    gemm_wmma_bf16<<<gEnc, 128, 0, stream>>>(
        h1b, Wencb, LSTM_D, DSLICES, DSLICES, 1,
        nullptr, nullptr, 0, 0,
        nullptr, b_enc, out + (size_t)t * LSTM_B * LSTM_V, LSTM_V, LSTM_V,
        prevb, VPAD);
  }
}
